// Net_91164975824992
// MI455X (gfx1250) — compile-verified, matrix operand order checked by reading, not run on hardware
//
#include <hip/hip_runtime.h>

#define NIN   784
#define HID   4096
#define NOUT  10
#define BATCH 256
#define TSTEPS 25
#define BETA  0.95f
#define THR   1.0f

#define KCH   128           // K-chunk staged per async transfer
#define NCH   (HID / KCH)   // 32 chunks
#define APAD  136           // 128 + 8 shorts: breaks LDS bank conflicts
#define SPAD  40            // 32 + 8 shorts

typedef __attribute__((ext_vector_type(16))) __bf16 v16bf;
typedef __attribute__((ext_vector_type(8)))  float  v8f;
typedef __attribute__((ext_vector_type(2)))  float  v2f;
typedef __attribute__((__vector_size__(16))) int    v4i;

typedef __attribute__((address_space(1))) v4i* gv4i_p;   // global v4i*
typedef __attribute__((address_space(3))) v4i* lv4i_p;   // LDS v4i*

static __device__ __forceinline__ unsigned short f32_to_bf16(float f) {
  unsigned u = __float_as_uint(f);
  unsigned r = 0x7FFFu + ((u >> 16) & 1u);   // round-to-nearest-even
  return (unsigned short)((u + r) >> 16);
}
static __device__ __forceinline__ float bf16_to_f32(unsigned short h) {
  return __uint_as_float(((unsigned)h) << 16);
}

union FragBF { uint4 u[2]; v16bf v; };

// 16-bit A(16x32)/B(32x16) fragment for lane (m16, half) from a row-major
// matrix row pointer: two contiguous b128 loads.
static __device__ __forceinline__ v16bf load_frag_bf16(const unsigned short* row,
                                                       int k0, int half) {
  FragBF f;
  f.u[0] = *(const uint4*)(row + k0 + 8 * half);        // K = 8h..8h+7   -> v0..v3
  f.u[1] = *(const uint4*)(row + k0 + 16 + 8 * half);   // K = 16+8h..+7  -> v4..v7
  return f.v;
}

// ---------------- prep: split W2 into bf16 hi/lo --------------------------
__global__ void convert_w2(const float* __restrict__ W2,
                           unsigned short* __restrict__ whi,
                           unsigned short* __restrict__ wlo) {
  size_t n = (size_t)HID * HID;
  size_t stride = (size_t)gridDim.x * blockDim.x;
  for (size_t i = (size_t)blockIdx.x * blockDim.x + threadIdx.x; i < n; i += stride) {
    float w = W2[i];
    unsigned short h = f32_to_bf16(w);
    whi[i] = h;
    wlo[i] = f32_to_bf16(w - bf16_to_f32(h));
  }
}

// ---------------- prep: W3 padded to 16 rows, bf16 hi/lo ------------------
__global__ void convert_w3(const float* __restrict__ W3,
                           unsigned short* __restrict__ w3hi,
                           unsigned short* __restrict__ w3lo) {
  int i = blockIdx.x * blockDim.x + threadIdx.x;   // [16][4096]
  if (i >= 16 * HID) return;
  int o = i >> 12;
  float w = (o < NOUT) ? W3[i] : 0.0f;             // rows 10..15 are zero pad
  unsigned short h = f32_to_bf16(w);
  w3hi[i] = h;
  w3lo[i] = f32_to_bf16(w - bf16_to_f32(h));
}

// ---------------- prep: zero mem2, out = b3 broadcast ---------------------
__global__ void init_misc(float* __restrict__ mem2, float* __restrict__ out,
                          const float* __restrict__ b3) {
  size_t i = (size_t)blockIdx.x * blockDim.x + threadIdx.x;
  if (i < (size_t)BATCH * HID) mem2[i] = 0.0f;
  if (i < (size_t)BATCH * TSTEPS * NOUT) out[i] = b3[i % NOUT];
}

// ---------------- prep: cur1 = relu(batch @ W1.T + b1); mem1_0; spk1(0) ---
__global__ void l1_kernel(const float* __restrict__ batch,
                          const float* __restrict__ W1,
                          const float* __restrict__ b1,
                          float* __restrict__ cur1,
                          float* __restrict__ mem1,
                          unsigned short* __restrict__ spk0) {
  const int lane = threadIdx.x & 31;
  const int wave = threadIdx.x >> 5;
  const int m16  = lane & 15;
  const int half = lane >> 4;
  const int bt = blockIdx.x & 15;       // 16 batch tiles
  const int nb = blockIdx.x >> 4;       // 32 n-blocks of 128
  const int b0 = bt * 16;
  const int n0 = nb * 128 + wave * 16;

  const float* arow = batch + (size_t)(b0 + m16) * NIN;
  const float* brow = W1    + (size_t)(n0 + m16) * NIN;   // B[k,n] = W1[n,k]

  v8f acc = {};
  #pragma unroll 4
  for (int k0 = 0; k0 < NIN; k0 += 4) {
    const float2 av = *(const float2*)(arow + k0 + 2 * half);
    const float2 bv = *(const float2*)(brow + k0 + 2 * half);
    v2f a, b;
    a[0] = av.x; a[1] = av.y;
    b[0] = bv.x; b[1] = bv.y;
    acc = __builtin_amdgcn_wmma_f32_16x16x4_f32(false, a, false, b,
                                                (short)0, acc, false, false);
  }

  const int col = n0 + m16;
  const float bias = b1[col];
  #pragma unroll
  for (int r = 0; r < 8; ++r) {
    const int row = b0 + r + 8 * half;
    float c = fmaxf(acc[r] + bias, 0.0f);
    size_t idx = (size_t)row * HID + col;
    cur1[idx] = c;
    mem1[idx] = c;                                     // beta*0 + cur1 - 0
    spk0[idx] = f32_to_bf16((c > THR) ? 1.0f : 0.0f);  // spk1 at t=0
  }
}

// ---------------- per-timestep fused kernel -------------------------------
__global__ void __launch_bounds__(256)
step_kernel(int t,
            const unsigned short* __restrict__ spk_in,
            unsigned short* __restrict__ spk_out,
            const unsigned short* __restrict__ whi,
            const unsigned short* __restrict__ wlo,
            const float* __restrict__ b2,
            const unsigned short* __restrict__ w3hi,
            const unsigned short* __restrict__ w3lo,
            const float* __restrict__ cur1,
            float* __restrict__ mem1,
            float* __restrict__ mem2,
            float* __restrict__ out) {
  __shared__ unsigned short a_stage[3][16][APAD];  // triple-buffered spk1 tile
  __shared__ unsigned short s2t[8][16][SPAD];      // per-wave spk2 tile (M x K)

  const int lane = threadIdx.x & 31;
  const int wave = threadIdx.x >> 5;
  const int m16  = lane & 15;
  const int half = lane >> 4;
  const int bt = blockIdx.x & 15;       // 16 batch tiles
  const int nb = blockIdx.x >> 4;       // 16 n-blocks of 256
  const int b0 = bt * 16;
  const int wgn0 = nb * 256;
  const int n0 = wgn0 + wave * 32;      // two 16-wide N tiles per wave

  const unsigned short* bh0 = whi + (size_t)(n0 + m16) * HID;
  const unsigned short* bh1 = whi + (size_t)(n0 + 16 + m16) * HID;
  const unsigned short* bl0 = wlo + (size_t)(n0 + m16) * HID;
  const unsigned short* bl1 = wlo + (size_t)(n0 + 16 + m16) * HID;

  // cooperative async staging: thread -> one 16B span of the 16x128 A chunk
  const int srow = threadIdx.x >> 4;
  const int sseg = threadIdx.x & 15;
  const unsigned short* gsrc = spk_in + (size_t)(b0 + srow) * HID + sseg * 8;
  unsigned short* ldst = &a_stage[0][srow][sseg * 8];

  auto stage = [&](int ch, int buf) {
    unsigned loff = (unsigned)(uintptr_t)(ldst) + (unsigned)(buf * 16 * APAD * 2);
    __builtin_amdgcn_global_load_async_to_lds_b128(
        (gv4i_p)(uintptr_t)(gsrc + ch * KCH),
        (lv4i_p)(uintptr_t)loff, 0, 0);
  };

  stage(0, 0);

  v8f acc0 = {}, acc1 = {};
  for (int ch = 0; ch < NCH; ++ch) {
    if (ch + 1 < NCH) {
      stage(ch + 1, (ch + 1) % 3);
      asm volatile("s_wait_asynccnt 0x1" ::: "memory");  // chunk ch has landed
    } else {
      asm volatile("s_wait_asynccnt 0x0" ::: "memory");
    }
    __syncthreads();

    const unsigned short* abuf = &a_stage[ch % 3][0][0];
    const int kg = ch * KCH;
    #pragma unroll
    for (int kk = 0; kk < KCH; kk += 32) {
      FragBF fa;
      fa.u[0] = *(const uint4*)(abuf + m16 * APAD + kk + 8 * half);
      fa.u[1] = *(const uint4*)(abuf + m16 * APAD + kk + 16 + 8 * half);
      v16bf a  = fa.v;
      v16bf h0 = load_frag_bf16(bh0, kg + kk, half);
      v16bf h1 = load_frag_bf16(bh1, kg + kk, half);
      v16bf l0 = load_frag_bf16(bl0, kg + kk, half);
      v16bf l1 = load_frag_bf16(bl1, kg + kk, half);
      acc0 = __builtin_amdgcn_wmma_f32_16x16x32_bf16(false, a, false, h0,
                                                     (short)0, acc0, false, false);
      acc1 = __builtin_amdgcn_wmma_f32_16x16x32_bf16(false, a, false, h1,
                                                     (short)0, acc1, false, false);
      acc0 = __builtin_amdgcn_wmma_f32_16x16x32_bf16(false, a, false, l0,
                                                     (short)0, acc0, false, false);
      acc1 = __builtin_amdgcn_wmma_f32_16x16x32_bf16(false, a, false, l1,
                                                     (short)0, acc1, false, false);
    }
  }

  // layer-2 leaky/spike; spk2 -> per-wave LDS tile in A-fragment order
  #pragma unroll
  for (int j = 0; j < 2; ++j) {
    const v8f accv = j ? acc1 : acc0;
    const int col = n0 + j * 16 + m16;
    const float bias = b2[col];
    #pragma unroll
    for (int r = 0; r < 8; ++r) {
      const int row = b0 + r + 8 * half;
      float c = fmaxf(accv[r] + bias, 0.0f);
      float* pm2 = mem2 + (size_t)row * HID + col;
      float m2 = *pm2;
      float reset = (m2 > THR) ? THR : 0.0f;
      m2 = BETA * m2 + c - reset;
      *pm2 = m2;
      s2t[wave][r + 8 * half][j * 16 + m16] =
          f32_to_bf16((m2 > THR) ? 1.0f : 0.0f);
    }
  }

  // layer-3 via WMMA: A = spk2 tile (16x32), B = W3pad cols (32x16 hi/lo)
  {
    FragBF fa;
    const unsigned short* trow = &s2t[wave][m16][0];
    fa.u[0] = *(const uint4*)(trow + 8 * half);
    fa.u[1] = *(const uint4*)(trow + 16 + 8 * half);
    v16bf a3 = fa.v;
    v16bf bh = load_frag_bf16(w3hi + (size_t)m16 * HID, n0, half);
    v16bf bl = load_frag_bf16(w3lo + (size_t)m16 * HID, n0, half);
    v8f acc3 = {};
    acc3 = __builtin_amdgcn_wmma_f32_16x16x32_bf16(false, a3, false, bh,
                                                   (short)0, acc3, false, false);
    acc3 = __builtin_amdgcn_wmma_f32_16x16x32_bf16(false, a3, false, bl,
                                                   (short)0, acc3, false, false);
    const int o = m16;                       // D: elem r of lane = (row r+8h, col o)
    if (o < NOUT) {
      #pragma unroll
      for (int r = 0; r < 8; ++r) {
        const int row = b0 + r + 8 * half;
        atomicAdd(out + (size_t)row * (TSTEPS * NOUT) + t * NOUT + o, acc3[r]);
      }
    }
  }

  // layer-1 leaky/spike for step t+1 (exclusive mem1 slice), vectorized
  {
    const int rrow = b0 + (threadIdx.x >> 4);
    const int c0 = wgn0 + (threadIdx.x & 15) * 16;
    const size_t base = (size_t)rrow * HID + c0;
    #pragma unroll
    for (int v = 0; v < 4; ++v) {
      float4 m1v = *(const float4*)(mem1 + base + v * 4);
      float4 c1v = *(const float4*)(cur1 + base + v * 4);
      float m[4] = { m1v.x, m1v.y, m1v.z, m1v.w };
      float c[4] = { c1v.x, c1v.y, c1v.z, c1v.w };
      unsigned pk[2];
      unsigned short s[4];
      #pragma unroll
      for (int e = 0; e < 4; ++e) {
        float reset = (m[e] > THR) ? THR : 0.0f;
        m[e] = BETA * m[e] + c[e] - reset;
        s[e] = f32_to_bf16((m[e] > THR) ? 1.0f : 0.0f);
      }
      pk[0] = (unsigned)s[0] | ((unsigned)s[1] << 16);
      pk[1] = (unsigned)s[2] | ((unsigned)s[3] << 16);
      *(float4*)(mem1 + base + v * 4) = make_float4(m[0], m[1], m[2], m[3]);
      *(uint2*)(spk_out + base + v * 4) = make_uint2(pk[0], pk[1]);
    }
  }
}

// --------------------------------------------------------------------------
extern "C" void kernel_launch(void* const* d_in, const int* in_sizes, int n_in,
                              void* d_out, int out_size, void* d_ws, size_t ws_size,
                              hipStream_t stream) {
  const float* batch = (const float*)d_in[0];
  const float* W1    = (const float*)d_in[1];
  const float* b1    = (const float*)d_in[2];
  const float* W2    = (const float*)d_in[3];
  const float* b2    = (const float*)d_in[4];
  const float* W3    = (const float*)d_in[5];
  const float* b3    = (const float*)d_in[6];
  float* out = (float*)d_out;

  char* ws = (char*)d_ws;                                             // ~80.3 MiB
  unsigned short* whi  = (unsigned short*)(ws);                       // 32 MiB
  unsigned short* wlo  = (unsigned short*)(ws + ((size_t)32 << 20));  // 32 MiB
  float* cur1          = (float*)(ws + ((size_t)64 << 20));           //  4 MiB
  float* mem1          = (float*)(ws + ((size_t)68 << 20));           //  4 MiB
  float* mem2          = (float*)(ws + ((size_t)72 << 20));           //  4 MiB
  unsigned short* spkA = (unsigned short*)(ws + ((size_t)76 << 20));  //  2 MiB
  unsigned short* spkB = (unsigned short*)(ws + ((size_t)78 << 20));  //  2 MiB
  unsigned short* w3hi = (unsigned short*)(ws + ((size_t)80 << 20));  // 128 KiB
  unsigned short* w3lo = (unsigned short*)(ws + ((size_t)80 << 20) + (1u << 17));

  hipLaunchKernelGGL(convert_w2, dim3(2048), dim3(256), 0, stream, W2, whi, wlo);
  hipLaunchKernelGGL(convert_w3, dim3(256),  dim3(256), 0, stream, W3, w3hi, w3lo);
  hipLaunchKernelGGL(init_misc,  dim3(4096), dim3(256), 0, stream, mem2, out, b3);
  hipLaunchKernelGGL(l1_kernel,  dim3(512),  dim3(256), 0, stream,
                     batch, W1, b1, cur1, mem1, spkA);

  unsigned short* bufs[2] = { spkA, spkB };
  for (int t = 0; t < TSTEPS; ++t) {
    hipLaunchKernelGGL(step_kernel, dim3(256), dim3(256), 0, stream, t,
                       bufs[t & 1], bufs[(t + 1) & 1],
                       whi, wlo, b2, w3hi, w3lo, cur1, mem1, mem2, out);
  }
}
